// InformedTestModel_33423435497942
// MI455X (gfx1250) — matrix operand order repaired and checked by reading, not compiled, not run on hardware
//
#include <hip/hip_runtime.h>

// ---------------------------------------------------------------------------
// InformedTestModel for MI455X (gfx1250): gather-embed + windowed rel-attn +
// flash causal MHA.  All matmuls use v_wmma_f32_16x16x32_f16 (wave32).
// B operands are staged transposed in LDS so every fragment read is a
// contiguous ds_load_b128 pair; A fragments are hoisted out of loops.
// Contiguous tile fills use GLOBAL_LOAD_ASYNC_TO_LDS (ASYNCcnt) and the MHA
// Q/K tiles use the Tensor Data Mover (TENSORcnt) when the builtins exist.
// ---------------------------------------------------------------------------

#define B_ 8
#define S_ 2048
#define NB_ (S_ / 64)  // 32 blocks of 64

typedef _Float16 half16 __attribute__((ext_vector_type(16)));
typedef float float8 __attribute__((ext_vector_type(8)));
typedef unsigned int u32x4 __attribute__((ext_vector_type(4)));
typedef int i32x4 __attribute__((ext_vector_type(4)));
typedef int i32x8 __attribute__((ext_vector_type(8)));
typedef int v4i __attribute__((vector_size(16)));  // GCC-style, for builtins

#if defined(__has_builtin)
#if __has_builtin(__builtin_amdgcn_tensor_load_to_lds) && \
    __has_builtin(__builtin_amdgcn_s_wait_tensorcnt)
#define HAVE_TDM 1
#endif
#if __has_builtin(__builtin_amdgcn_global_load_async_to_lds_b128) && \
    __has_builtin(__builtin_amdgcn_s_wait_asynccnt)
#define HAVE_ASYNC_LDS 1
#endif
#endif
#ifndef HAVE_TDM
#define HAVE_TDM 0
#endif
#ifndef HAVE_ASYNC_LDS
#define HAVE_ASYNC_LDS 0
#endif

__device__ __forceinline__ float8 zero8() {
  float8 z = {0.f, 0.f, 0.f, 0.f, 0.f, 0.f, 0.f, 0.f};
  return z;
}

__device__ __forceinline__ float8 wmma_f16(half16 a, half16 b, float8 c) {
  // (neg_a, A, neg_b, B, c_mod, C, reuse_a, reuse_b)
  return __builtin_amdgcn_wmma_f32_16x16x32_f16(false, a, false, b, (short)0, c,
                                                false, false);
}

// A fragment (16x32, MxK) from a row-major matrix.
// Lane l: m = l&15, half = l>>4; element j -> K = j + 16*half.
__device__ __forceinline__ half16 frag_a(const _Float16* base, int ld, int row0,
                                         int col0) {
  const int lane = threadIdx.x & 31;
  const int m = lane & 15;
  const int hh = lane >> 4;
  const _Float16* p = base + (size_t)(row0 + m) * ld + col0 + 16 * hh;
  half16 a;
#pragma unroll
  for (int j = 0; j < 8; ++j) a[j] = p[j];
#pragma unroll
  for (int j = 0; j < 8; ++j) a[j + 8] = p[j + 8];
  return a;
}

// B fragment (32x16, KxN) where memory holds the TRANSPOSE (N rows x K cols,
// row-major).  Lane l: n = l&15, half = l>>4; element j -> K = j + 16*half.
// Contiguous 32B per lane -> two ds_load_b128.
__device__ __forceinline__ half16 frag_bt(const _Float16* base, int ld, int k0,
                                          int n0) {
  const int lane = threadIdx.x & 31;
  const int nn = lane & 15;
  const int hh = lane >> 4;
  const _Float16* p = base + (size_t)(n0 + nn) * ld + k0 + 16 * hh;
  half16 b;
#pragma unroll
  for (int j = 0; j < 16; ++j) b[j] = p[j];
  return b;
}

// Scatter one uint4 (8 f16 of row r, cols c..c+7) into a transposed LDS tile.
__device__ __forceinline__ void store_transposed8(_Float16* sT, int ldT, int r,
                                                  int c, uint4 v) {
  unsigned short* st = (unsigned short*)sT;
  const unsigned w[4] = {v.x, v.y, v.z, v.w};
#pragma unroll
  for (int x = 0; x < 4; ++x) {
    st[(size_t)(c + 2 * x) * ldT + r] = (unsigned short)(w[x] & 0xffffu);
    st[(size_t)(c + 2 * x + 1) * ldT + r] = (unsigned short)(w[x] >> 16);
  }
}

#if HAVE_ASYNC_LDS
// 16B per lane global -> LDS, tracked on ASYNCcnt (no VGPR round trip).
// Builtin expects v4i32 pointers: (v4i addrspace(1)*, v4i addrspace(3)*,
// imm offset, imm cpol).
__device__ __forceinline__ void async_copy_b128(const _Float16* gsrc,
                                                _Float16* ldst) {
  __builtin_amdgcn_global_load_async_to_lds_b128(
      (__attribute__((address_space(1))) v4i*)(size_t)gsrc,
      (__attribute__((address_space(3))) v4i*)(unsigned)(size_t)ldst,
      0, 0);
}
#endif

#if HAVE_TDM
// Issue a 2-D f16 tile load through the Tensor Data Mover.
// D# per CDNA5 ISA 8.3/8.4: group0 = {count|flags, lds_addr, global_addr,
// type=2}; group1 = {data_size, tensor dims, tile dims, stride}.
__device__ __forceinline__ void tdm_load_2d_f16(const _Float16* gsrc,
                                                _Float16* ldst, int rows,
                                                int cols, int stride) {
  const unsigned lds_off = (unsigned)(size_t)(void*)ldst;
  const unsigned long long ga = (unsigned long long)(size_t)gsrc;
  u32x4 g0;
  g0[0] = 1u;                              // count=1, user mode
  g0[1] = lds_off;                         // lds_addr (bytes)
  g0[2] = (unsigned)(ga & 0xffffffffu);    // global_addr[31:0]
  g0[3] = (unsigned)((ga >> 32) & 0x1ffffffu) | (2u << 30);  // type=2
  i32x8 g1;
  g1[0] = 0x00010000;                        // data_size=1 -> 2 bytes/elem
  g1[1] = (int)((unsigned)stride << 16);     // tensor_dim0 low16 (bits 63:48)
  g1[2] = (int)((unsigned)rows << 16);       // tensor_dim1 low16 (bits 95:80)
  g1[3] = (int)((unsigned)cols << 16);       // tile_dim0 (bits 127:112)
  g1[4] = rows;                              // tile_dim1 (bits 143:128)
  g1[5] = stride;                            // tensor_dim0_stride low32
  g1[6] = 0;
  g1[7] = 0;
  i32x4 gz = {0, 0, 0, 0};
#if __clang_major__ >= 23
  i32x8 gz8 = {0, 0, 0, 0, 0, 0, 0, 0};
  __builtin_amdgcn_tensor_load_to_lds(g0, g1, gz, gz, gz8, 0);
#else
  __builtin_amdgcn_tensor_load_to_lds(g0, g1, gz, gz, 0);
#endif
}
#endif

// ---------------------------------------------------------------------------
// 1) one_hot @ W  ==  row gather (f32 table -> f16 activations)
// ---------------------------------------------------------------------------
__global__ __launch_bounds__(128) void gather_embed_kernel(
    const int* __restrict__ xs, const float* __restrict__ lq,
    const float* __restrict__ lk, const float* __restrict__ lv,
    _Float16* __restrict__ qL, _Float16* __restrict__ kL,
    _Float16* __restrict__ vL) {
  const int bs = blockIdx.x;  // 0 .. B*S-1
  const int idx = xs[bs];
  const int t = threadIdx.x;
  const float qscale = 0.0625f;  // 1/sqrt(256)
#pragma unroll
  for (int i = 0; i < 2; ++i) {
    const int c = t * 2 + i;
    qL[(size_t)bs * 256 + c] = (_Float16)(lq[idx * 256 + c] * qscale);
    kL[(size_t)bs * 256 + c] = (_Float16)(lk[idx * 256 + c]);
  }
#pragma unroll
  for (int i = 0; i < 4; ++i) {
    const int c = t * 4 + i;
    vL[(size_t)bs * 512 + c] = (_Float16)(lv[idx * 512 + c]);
  }
}

// f32 -> f16 weight conversion (optionally pre-scaled)
__global__ void cvt_kernel(const float* __restrict__ src,
                           _Float16* __restrict__ dst, int n, float scale) {
  const int i = blockIdx.x * blockDim.x + threadIdx.x;
  if (i < n) dst[i] = (_Float16)(src[i] * scale);
}

// ---------------------------------------------------------------------------
// 2) Windowed relative attention + ReLU.  One workgroup per (b, block).
// ---------------------------------------------------------------------------
__global__ __launch_bounds__(256) void local_attn_kernel(
    const _Float16* __restrict__ qL, const _Float16* __restrict__ kL,
    const _Float16* __restrict__ vL, const float* __restrict__ rel_bias,
    _Float16* __restrict__ xmid) {
  __shared__ _Float16 sQ[64 * 256];    // 32KB; reused for probs (64x192 f16)
  __shared__ _Float16 sKV[192 * 128];  // 48KB; K chunk row-major or V^T chunk
  __shared__ float sS[64 * 192];       // 48KB scores

  const int b = blockIdx.x / NB_;
  const int n = blockIdx.x % NB_;
  const int tid = threadIdx.x;
  const int wave = tid >> 5;
  const int lane = tid & 31;
  const int mt = wave >> 1;    // fixed 16-row band per wave
  const int ntb = (wave & 1);  // column-half selector

  // ---- load Q tile (64x256 f16, contiguous) ----
  {
    const _Float16* g = qL + ((size_t)b * S_ + n * 64) * 256;
#if HAVE_ASYNC_LDS
    for (int i = tid; i < 64 * 256 / 8; i += 256)
      async_copy_b128(g + (size_t)i * 8, sQ + (size_t)i * 8);
    __builtin_amdgcn_s_wait_asynccnt(0);
#else
    const uint4* src = (const uint4*)g;
    uint4* dst = (uint4*)sQ;
    for (int i = tid; i < 64 * 256 / 8; i += 256) dst[i] = src[i];
#endif
  }
  __syncthreads();

  // ---- phase 1: scores = Q * K^T per neighbor block ----
  // Hoist A fragments: invariant across the 3 key blocks.
  half16 qa[8];
#pragma unroll
  for (int kc = 0; kc < 8; ++kc) qa[kc] = frag_a(sQ, 256, mt * 16, kc * 32);

  for (int kb = 0; kb < 3; ++kb) {
    const int kn = n - 1 + kb;
    const bool valid = (kn >= 0) && (kn < NB_);
    __syncthreads();  // previous iteration's consumers done
    if (valid) {
      const _Float16* g = kL + ((size_t)b * S_ + kn * 64) * 256;
#if HAVE_ASYNC_LDS
      for (int i = tid; i < 64 * 256 / 8; i += 256)
        async_copy_b128(g + (size_t)i * 8, sKV + (size_t)i * 8);
#else
      const uint4* src = (const uint4*)g;
      uint4* dst = (uint4*)sKV;
      for (int i = tid; i < 64 * 256 / 8; i += 256) dst[i] = src[i];
#endif
    } else {
      uint4* dst = (uint4*)sKV;
      uint4 z;
      z.x = 0u; z.y = 0u; z.z = 0u; z.w = 0u;
      for (int i = tid; i < 64 * 256 / 8; i += 256) dst[i] = z;
    }
#if HAVE_ASYNC_LDS
    __builtin_amdgcn_s_wait_asynccnt(0);
#endif
    __syncthreads();
#pragma unroll
    for (int tt = 0; tt < 2; ++tt) {
      const int nt = ntb * 2 + tt;
      float8 acc = zero8();
#pragma unroll
      for (int kc = 0; kc < 8; ++kc)
        acc = wmma_f16(qa[kc], frag_bt(sKV, 256, kc * 32, nt * 16), acc);
      const int nn = lane & 15, hh = lane >> 4;
#pragma unroll
      for (int r = 0; r < 8; ++r)
        sS[(mt * 16 + r + 8 * hh) * 192 + kb * 64 + nt * 16 + nn] = acc[r];
    }
  }
  __syncthreads();

  // ---- phase 2: relative bias + window mask + softmax ----
  _Float16* sP = sQ;  // reuse Q region for f16 probabilities (64x192)
  for (int rr = 0; rr < 8; ++rr) {
    const int row = wave * 8 + rr;
    float vals[6];
    float vmax = -3.0e38f;
#pragma unroll
    for (int i = 0; i < 6; ++i) {
      const int c = lane * 6 + i;      // 32 lanes * 6 = 192 cols
      const int rel = (c - 64) - row;  // kcol - qpos
      const int gpos = n * 64 + (c - 64);
      float s = sS[row * 192 + c];
      const bool ok = (rel >= -64) && (rel <= 64) && (gpos >= 0) && (gpos < S_);
      s = ok ? (s + rel_bias[rel + 64]) : -1.0e9f;
      vals[i] = s;
      vmax = fmaxf(vmax, s);
    }
#pragma unroll
    for (int off = 16; off > 0; off >>= 1)
      vmax = fmaxf(vmax, __shfl_xor(vmax, off, 32));
    float sum = 0.f;
#pragma unroll
    for (int i = 0; i < 6; ++i) {
      vals[i] = __expf(vals[i] - vmax);
      sum += vals[i];
    }
#pragma unroll
    for (int off = 16; off > 0; off >>= 1) sum += __shfl_xor(sum, off, 32);
    const float inv = 1.f / sum;
#pragma unroll
    for (int i = 0; i < 6; ++i)
      sP[row * 192 + lane * 6 + i] = (_Float16)(vals[i] * inv);
  }
  __syncthreads();

  // ---- phase 3: O = P(64x192) * V(192x512), 4 chunks of 128 cols ----
  // Hoist P fragments: invariant across the 4 column chunks.
  half16 pa[6];
#pragma unroll
  for (int kc = 0; kc < 6; ++kc) pa[kc] = frag_a(sP, 192, mt * 16, kc * 32);

  for (int cc = 0; cc < 4; ++cc) {
    __syncthreads();
    // V chunk stored TRANSPOSED: sVt[col][row], col 0..127, row 0..191.
    for (int i = tid; i < 192 * 128 / 8; i += 256) {
      const int e = i * 8;
      const int r = e / 128;  // 0..191 (key row)
      const int c = e % 128;  // col within chunk
      const int kb = r / 64;
      const int rr2 = r % 64;
      const int kn = n - 1 + kb;
      uint4 v;
      if (kn >= 0 && kn < NB_) {
        v = *(const uint4*)(vL + ((size_t)b * S_ + kn * 64 + rr2) * 512 +
                            cc * 128 + c);
      } else {
        v.x = 0u; v.y = 0u; v.z = 0u; v.w = 0u;
      }
      store_transposed8(sKV, 192, r, c, v);
    }
    __syncthreads();
#pragma unroll
    for (int tt = 0; tt < 4; ++tt) {
      const int nt = ntb * 4 + tt;  // 8 col tiles per chunk
      float8 acc = zero8();
#pragma unroll
      for (int kc = 0; kc < 6; ++kc)
        acc = wmma_f16(pa[kc], frag_bt(sKV, 192, kc * 32, nt * 16), acc);
      const int nn = lane & 15, hh = lane >> 4;
#pragma unroll
      for (int r = 0; r < 8; ++r) {
        const int row = mt * 16 + r + 8 * hh;
        const int col = cc * 128 + nt * 16 + nn;
        const float v = fmaxf(acc[r], 0.f);  // ReLU
        xmid[((size_t)b * S_ + n * 64 + row) * 512 + col] = (_Float16)v;
      }
    }
  }
}

// ---------------------------------------------------------------------------
// 3/5) Generic WMMA GEMM: C[MxN] = A[MxK] * B[KxN], f16 in, f16 or f32 out.
//      B tile is staged transposed in LDS.
// ---------------------------------------------------------------------------
template <bool OUT_F32>
__global__ __launch_bounds__(256) void gemm_kernel(
    const _Float16* __restrict__ A, const _Float16* __restrict__ Bm,
    void* __restrict__ C, int M, int N, int K) {
  __shared__ _Float16 sA[64 * 64];
  __shared__ _Float16 sBt[64 * 64];  // B^T: [n][k]
  const int tid = threadIdx.x;
  const int wave = tid >> 5;
  const int lane = tid & 31;
  const int nBlk = blockIdx.x;
  const int mBlk = blockIdx.y;
  const int mt = wave >> 1;
  const int ntb = (wave & 1);
  float8 acc[2];
  acc[0] = zero8();
  acc[1] = zero8();

  for (int k0 = 0; k0 < K; k0 += 64) {
    {  // A: 64 rows x 8 uint4 (contiguous per row)
      const _Float16* g = A + (size_t)(mBlk * 64) * K + k0;
      for (int i = tid; i < 512; i += 256) {
        const int r = i >> 3;
        const int c = (i & 7) * 8;
#if HAVE_ASYNC_LDS
        async_copy_b128(g + (size_t)r * K + c, sA + r * 64 + c);
#else
        *(uint4*)(sA + r * 64 + c) = *(const uint4*)(g + (size_t)r * K + c);
#endif
      }
    }
    for (int i = tid; i < 512; i += 256) {  // B: 64 k-rows x 8 uint4 -> B^T
      const int r = i >> 3;        // k index within tile
      const int c8 = (i & 7) * 8;  // col within tile (multiple of 8)
      const int col = nBlk * 64 + c8;
      uint4 v;
      if (col < N) {
        v = *(const uint4*)(Bm + (size_t)(k0 + r) * N + col);
      } else {
        v.x = 0u; v.y = 0u; v.z = 0u; v.w = 0u;
      }
      store_transposed8(sBt, 64, r, c8, v);
    }
#if HAVE_ASYNC_LDS
    __builtin_amdgcn_s_wait_asynccnt(0);
#endif
    __syncthreads();
    half16 a2[2];
#pragma unroll
    for (int kc = 0; kc < 2; ++kc) a2[kc] = frag_a(sA, 64, mt * 16, kc * 32);
#pragma unroll
    for (int tt = 0; tt < 2; ++tt) {
      const int nt = ntb * 2 + tt;
#pragma unroll
      for (int kc = 0; kc < 2; ++kc)
        acc[tt] = wmma_f16(a2[kc], frag_bt(sBt, 64, kc * 32, nt * 16), acc[tt]);
    }
    __syncthreads();
  }

  const int nn = lane & 15, hh = lane >> 4;
#pragma unroll
  for (int tt = 0; tt < 2; ++tt) {
    const int nt = ntb * 2 + tt;
#pragma unroll
    for (int r = 0; r < 8; ++r) {
      const int row = mBlk * 64 + mt * 16 + r + 8 * hh;
      const int col = nBlk * 64 + nt * 16 + nn;
      if (col < N) {
        if (OUT_F32)
          ((float*)C)[(size_t)row * N + col] = acc[tt][r];
        else
          ((_Float16*)C)[(size_t)row * N + col] = (_Float16)acc[tt][r];
      }
    }
  }
}

// ---------------------------------------------------------------------------
// 4) Causal MHA, flash-attention style.  One workgroup per (qblock, h, b).
//    dk=32 per head -> one WMMA per 16x16 score tile.  dv=128.
//    Q/K tiles fetched by the Tensor Data Mover when available.
// ---------------------------------------------------------------------------
__global__ __launch_bounds__(256) void causal_mha_kernel(
    const _Float16* __restrict__ qp, const _Float16* __restrict__ kp,
    const _Float16* __restrict__ vp, _Float16* __restrict__ op) {
  __shared__ _Float16 sQ[64 * 32];
  __shared__ _Float16 sK[64 * 32];    // row-major (used via frag_bt)
  __shared__ _Float16 sVt[128 * 64];  // V^T: [dv col][key row]
  __shared__ float sS[64 * 64];
  __shared__ _Float16 sP[64 * 64];
  __shared__ float sMrow[64];
  __shared__ float sLrow[64];
  __shared__ float sAl[64];

  const int qb = blockIdx.x;
  const int h = blockIdx.y;
  const int b = blockIdx.z;
  const int tid = threadIdx.x;
  const int wave = tid >> 5;
  const int lane = tid & 31;
  const int mt = wave >> 1;
  const int ntb = (wave & 1);

  {  // Q 64x32 (row stride 256)
    const _Float16* g = qp + ((size_t)b * S_ + qb * 64) * 256 + h * 32;
#if HAVE_TDM
    if (wave == 0) tdm_load_2d_f16(g, sQ, 64, 32, 256);
#else
    for (int i = tid; i < 64 * 32 / 8; i += 256) {
      const int r = i >> 2;
      const int c = (i & 3) * 8;
#if HAVE_ASYNC_LDS
      async_copy_b128(g + (size_t)r * 256 + c, sQ + r * 32 + c);
#else
      *(uint4*)(sQ + r * 32 + c) = *(const uint4*)(g + (size_t)r * 256 + c);
#endif
    }
#endif
  }
  if (tid < 64) {
    sMrow[tid] = -3.0e38f;
    sLrow[tid] = 0.f;
  }
  float8 acc[4];
#pragma unroll
  for (int tt = 0; tt < 4; ++tt) acc[tt] = zero8();
#if HAVE_TDM
  if (wave == 0) __builtin_amdgcn_s_wait_tensorcnt(0);
#elif HAVE_ASYNC_LDS
  __builtin_amdgcn_s_wait_asynccnt(0);
#endif
  __syncthreads();

  // Q A-fragment: invariant across the whole key loop (single K step, dk=32).
  const half16 qa = frag_a(sQ, 32, mt * 16, 0);

  for (int j = 0; j <= qb; ++j) {
    __syncthreads();  // previous iteration fully consumed
    // ---- K tile: 64x32 f16, row stride 256 elements ----
    {
      const _Float16* g = kp + ((size_t)b * S_ + j * 64) * 256 + h * 32;
#if HAVE_TDM
      if (wave == 0) tdm_load_2d_f16(g, sK, 64, 32, 256);
#else
      for (int i = tid; i < 64 * 32 / 8; i += 256) {
        const int r = i >> 2;
        const int c = (i & 3) * 8;
#if HAVE_ASYNC_LDS
        async_copy_b128(g + (size_t)r * 256 + c, sK + r * 32 + c);
#else
        *(uint4*)(sK + r * 32 + c) = *(const uint4*)(g + (size_t)r * 256 + c);
#endif
      }
#endif
    }
    // ---- V tile 64x128, stored transposed (row stride 1024 elements) ----
    for (int i = tid; i < 64 * 128 / 8; i += 256) {
      const int r = i >> 4;
      const int c = (i & 15) * 8;
      uint4 v = *(const uint4*)(vp + ((size_t)b * S_ + j * 64 + r) * 1024 +
                                h * 128 + c);
      store_transposed8(sVt, 64, r, c, v);
    }
    // prefetch next block's K/V lines while this block computes
    if (j < qb) {
      const _Float16* kn = kp + ((size_t)b * S_ + (j + 1) * 64) * 256 + h * 32;
      const _Float16* vn = vp + ((size_t)b * S_ + (j + 1) * 64) * 1024 + h * 128;
      __builtin_prefetch(kn + (size_t)(tid & 63) * 256, 0, 0);
      __builtin_prefetch(vn + (size_t)(tid & 63) * 1024, 0, 0);
    }
#if HAVE_TDM
    if (wave == 0) __builtin_amdgcn_s_wait_tensorcnt(0);
#elif HAVE_ASYNC_LDS
    __builtin_amdgcn_s_wait_asynccnt(0);
#endif
    __syncthreads();

    // ---- scores (scale folded into Wq) ----
#pragma unroll
    for (int tt = 0; tt < 2; ++tt) {
      const int nt = ntb * 2 + tt;
      float8 s = wmma_f16(qa, frag_bt(sK, 32, 0, nt * 16), zero8());
      const int nn = lane & 15, hh = lane >> 4;
#pragma unroll
      for (int r = 0; r < 8; ++r)
        sS[(mt * 16 + r + 8 * hh) * 64 + nt * 16 + nn] = s[r];
    }
    __syncthreads();

    // ---- online softmax: 8 rows per wave, 2 cols per lane ----
    for (int rr = 0; rr < 8; ++rr) {
      const int row = wave * 8 + rr;
      const int qg = qb * 64 + row;
      float vals[2];
      float vmax = -3.0e38f;
#pragma unroll
      for (int i = 0; i < 2; ++i) {
        const int c = lane * 2 + i;
        const int kg = j * 64 + c;
        float s = sS[row * 64 + c];
        s = (kg <= qg) ? s : -1.0e9f;
        vals[i] = s;
        vmax = fmaxf(vmax, s);
      }
#pragma unroll
      for (int off = 16; off > 0; off >>= 1)
        vmax = fmaxf(vmax, __shfl_xor(vmax, off, 32));
      const float mold = sMrow[row];
      const float mnew = fmaxf(mold, vmax);
      const float alpha = __expf(mold - mnew);
      float sum = 0.f;
#pragma unroll
      for (int i = 0; i < 2; ++i) {
        vals[i] = __expf(vals[i] - mnew);
        sum += vals[i];
      }
#pragma unroll
      for (int off = 16; off > 0; off >>= 1) sum += __shfl_xor(sum, off, 32);
#pragma unroll
      for (int i = 0; i < 2; ++i)
        sP[row * 64 + lane * 2 + i] = (_Float16)vals[i];
      if (lane == 0) {
        sAl[row] = alpha;
        sMrow[row] = mnew;
        sLrow[row] = sLrow[row] * alpha + sum;
      }
    }
    __syncthreads();

    // ---- rescale running O and accumulate P*V ----
    half16 pa[2];
#pragma unroll
    for (int kc = 0; kc < 2; ++kc) pa[kc] = frag_a(sP, 64, mt * 16, kc * 32);
    const int hh = lane >> 4;
#pragma unroll
    for (int tt = 0; tt < 4; ++tt) {
      const int nt = ntb * 4 + tt;
#pragma unroll
      for (int r = 0; r < 8; ++r) acc[tt][r] *= sAl[mt * 16 + r + 8 * hh];
#pragma unroll
      for (int kc = 0; kc < 2; ++kc)
        acc[tt] = wmma_f16(pa[kc], frag_bt(sVt, 64, kc * 32, nt * 16), acc[tt]);
    }
  }

  // ---- normalize and store f16 output [B, S, H*128] ----
  const int nn = lane & 15, hh = lane >> 4;
#pragma unroll
  for (int tt = 0; tt < 4; ++tt) {
    const int nt = ntb * 4 + tt;
#pragma unroll
    for (int r = 0; r < 8; ++r) {
      const int row = mt * 16 + r + 8 * hh;
      const float v = acc[tt][r] / sLrow[row];
      op[((size_t)b * S_ + qb * 64 + row) * 1024 + h * 128 + nt * 16 + nn] =
          (_Float16)v;
    }
  }
}

// ---------------------------------------------------------------------------
extern "C" void kernel_launch(void* const* d_in, const int* in_sizes, int n_in,
                              void* d_out, int out_size, void* d_ws,
                              size_t ws_size, hipStream_t stream) {
  (void)in_sizes; (void)n_in; (void)out_size; (void)ws_size;
  const int* xs = (const int*)d_in[0];
  const float* lq = (const float*)d_in[1];
  const float* lk = (const float*)d_in[2];
  const float* lv = (const float*)d_in[3];
  const float* relb = (const float*)d_in[4];
  const float* pq = (const float*)d_in[5];
  const float* pk = (const float*)d_in[6];
  const float* pv = (const float*)d_in[7];
  const float* po = (const float*)d_in[8];

  _Float16* w = (_Float16*)d_ws;
  size_t off = 0;
  auto alloc = [&](size_t nelem) {
    _Float16* p = w + off;
    off += nelem;
    return p;
  };
  _Float16* qL = alloc((size_t)B_ * S_ * 256);
  _Float16* kL = alloc((size_t)B_ * S_ * 256);
  _Float16* vL = alloc((size_t)B_ * S_ * 512);
  _Float16* xmid = alloc((size_t)B_ * S_ * 512);
  _Float16* wq = alloc(512 * 256);
  _Float16* wk = alloc(512 * 256);
  _Float16* wv = alloc(512 * 1024);
  _Float16* wo = alloc(1024 * 240);
  _Float16* qproj = alloc((size_t)B_ * S_ * 256);
  _Float16* kproj = alloc((size_t)B_ * S_ * 256);
  _Float16* vproj = alloc((size_t)B_ * S_ * 1024);
  _Float16* oattn = alloc((size_t)B_ * S_ * 1024);

  gather_embed_kernel<<<B_ * S_, 128, 0, stream>>>(xs, lq, lk, lv, qL, kL, vL);

  cvt_kernel<<<(512 * 256 + 255) / 256, 256, 0, stream>>>(
      pq, wq, 512 * 256, 0.17677669529663687f);  // 1/sqrt(32)
  cvt_kernel<<<(512 * 256 + 255) / 256, 256, 0, stream>>>(pk, wk, 512 * 256,
                                                          1.f);
  cvt_kernel<<<(512 * 1024 + 255) / 256, 256, 0, stream>>>(pv, wv, 512 * 1024,
                                                           1.f);
  cvt_kernel<<<(1024 * 240 + 255) / 256, 256, 0, stream>>>(po, wo, 1024 * 240,
                                                           1.f);

  local_attn_kernel<<<B_ * NB_, 256, 0, stream>>>(qL, kL, vL, relb, xmid);

  const int M = B_ * S_;
  gemm_kernel<false><<<dim3(256 / 64, M / 64), 256, 0, stream>>>(
      xmid, wq, qproj, M, 256, 512);
  gemm_kernel<false><<<dim3(256 / 64, M / 64), 256, 0, stream>>>(
      xmid, wk, kproj, M, 256, 512);
  gemm_kernel<false><<<dim3(1024 / 64, M / 64), 256, 0, stream>>>(
      xmid, wv, vproj, M, 1024, 512);

  causal_mha_kernel<<<dim3(S_ / 64, 8, B_), 256, 0, stream>>>(qproj, kproj,
                                                              vproj, oattn);

  gemm_kernel<true><<<dim3((240 + 63) / 64, M / 64), 256, 0, stream>>>(
      oattn, wo, d_out, M, 240, 1024);
}
